// Hamburger_79637283602491
// MI455X (gfx1250) — matrix operand order repaired and checked by reading, not compiled, not run on hardware
//
#include <hip/hip_runtime.h>
#include <hip/hip_bf16.h>
#include <math.h>

typedef __attribute__((ext_vector_type(16))) _Float16 v16h;
typedef __attribute__((ext_vector_type(8)))  _Float16 v8h;
typedef __attribute__((ext_vector_type(2)))  _Float16 v2h;
typedef __attribute__((ext_vector_type(8)))  float    v8f;

#define NMF_EPS 1e-6f
#define LS 40   // LDS tile stride in halfs (80B: 16B-aligned, bank-skewed)

// ---------------------------------------------------------------------------
// Fragment loads (wave32, ISA 7.12.2 layouts), contiguous -> ds_load_b128
// A tile in LDS row-major As[m][k] (stride LS):
//   lane 0-15: M=lane, K={hs*8..+7, 16+hs*8..+7} with hs=0 ; lanes 16-31: hs=1
__device__ __forceinline__ v16h load_a_frag(const _Float16* As, int mo, int lane) {
  const _Float16* p = As + (mo + (lane & 15)) * LS + ((lane >> 4) << 3);
  v8h lo = *(const v8h*)(p);        // K = hs*8 .. hs*8+7
  v8h hi = *(const v8h*)(p + 16);   // K = 16+hs*8 .. +7
  return __builtin_shufflevector(lo, hi, 0,1,2,3,4,5,6,7,8,9,10,11,12,13,14,15);
}
// B tile in LDS column-major Bs[n][k] (stride LS):
//   lane 0-15: N=lane, K={hs*8..+7, 16+hs*8..+7} with hs=0 ; lanes 16-31: hs=1
__device__ __forceinline__ v16h load_b_frag(const _Float16* Bs, int no, int lane) {
  const _Float16* p = Bs + (no + (lane & 15)) * LS + ((lane >> 4) << 3);
  v8h lo = *(const v8h*)(p);
  v8h hi = *(const v8h*)(p + 16);
  return __builtin_shufflevector(lo, hi, 0,1,2,3,4,5,6,7,8,9,10,11,12,13,14,15);
}

// Stage a 64x32 f32->f16 A tile into As[m][k]; pairs of consecutive k packed.
// TA: A[m][k] = Ab[k*lda + m]  (coalesce over m) ; else coalesce over k.
template<bool TA>
__device__ __forceinline__ void stage_a(const float* __restrict__ A, long lda,
                                        int m0, int k0, int tid, _Float16* As) {
#pragma unroll
  for (int j = 0; j < 4; ++j) {
    int p = tid + j * 256;            // 1024 k-pairs: 64 m x 16 pairs
    int m, kp;
    if (TA) { m = p & 63; kp = p >> 6; }
    else    { kp = p & 15; m = p >> 4; }
    int k = k0 + 2 * kp;
    float v0, v1;
    if (TA) { v0 = A[(long)k * lda + (m0 + m)]; v1 = A[(long)(k + 1) * lda + (m0 + m)]; }
    else    { const float* ap = A + (long)(m0 + m) * lda + k; v0 = ap[0]; v1 = ap[1]; }
    v2h pk = { (_Float16)v0, (_Float16)v1 };
    *(v2h*)(As + m * LS + 2 * kp) = pk;
  }
}
// Stage a 32x64 B tile into Bs[n][k] (column-major).
// TB: B[k][n] = Bb[n*ldb + k] (coalesce over k) ; else coalesce over n.
template<bool TB>
__device__ __forceinline__ void stage_b(const float* __restrict__ B, long ldb,
                                        int n0, int k0, int tid, _Float16* Bs) {
#pragma unroll
  for (int j = 0; j < 4; ++j) {
    int p = tid + j * 256;
    int n, kp;
    if (TB) { kp = p & 15; n = p >> 4; }
    else    { n = p & 63; kp = p >> 6; }
    int k = k0 + 2 * kp;
    float v0, v1;
    if (TB) { const float* bp = B + (long)(n0 + n) * ldb + k; v0 = bp[0]; v1 = bp[1]; }
    else    { v0 = B[(long)k * ldb + (n0 + n)]; v1 = B[(long)(k + 1) * ldb + (n0 + n)]; }
    v2h pk = { (_Float16)v0, (_Float16)v1 };
    *(v2h*)(Bs + n * LS + 2 * kp) = pk;
  }
}

// ---------------------------------------------------------------------------
// Generic batched GEMM: C[b] = op(A[b]) * op(B[b]) (+bias, relu)
// Block tile 64x64, 256 threads = 8 waves, 2 16x16 output tiles per wave.
template<bool TA, bool TB>
__global__ void __launch_bounds__(256) gemm_wmma_kernel(
    const float* __restrict__ A, const float* __restrict__ B, float* __restrict__ C,
    int K, int lda, int ldb, int ldc,
    long sA, long sB, long sC,
    const float* __restrict__ bias, int relu)
{
  __shared__ _Float16 As[64 * LS];
  __shared__ _Float16 Bs[64 * LS];
  const int bz = blockIdx.z;
  A += (long)bz * sA; B += (long)bz * sB; C += (long)bz * sC;
  const int m0 = blockIdx.y * 64;
  const int n0 = blockIdx.x * 64;
  const int tid = threadIdx.x;
  const int wave = tid >> 5, lane = tid & 31;
  const int tr = (wave & 3) * 16;        // m offset of this wave's tile row
  const int tc = (wave >> 2) * 32;       // n offset (two adjacent tiles)
  v8f acc0 = {}, acc1 = {};

  for (int k0 = 0; k0 < K; k0 += 32) {
    stage_a<TA>(A, lda, m0, k0, tid, As);
    stage_b<TB>(B, ldb, n0, k0, tid, Bs);
    __syncthreads();
    v16h af = load_a_frag(As, tr, lane);
    v16h b0 = load_b_frag(Bs, tc, lane);
    v16h b1 = load_b_frag(Bs, tc + 16, lane);
    acc0 = __builtin_amdgcn_wmma_f32_16x16x32_f16(false, af, false, b0, (short)0, acc0, false, false);
    acc1 = __builtin_amdgcn_wmma_f32_16x16x32_f16(false, af, false, b1, (short)0, acc1, false, false);
    __syncthreads();
  }
  const int ncol = n0 + tc + (lane & 15);
  const int mr   = m0 + tr + ((lane >> 4) * 8);
#pragma unroll
  for (int i = 0; i < 8; ++i) {
    const int m = mr + i;
    float v0 = acc0[i], v1 = acc1[i];
    if (bias) { float bb = bias[m]; v0 += bb; v1 += bb; }
    if (relu) { v0 = fmaxf(v0, 0.f); v1 = fmaxf(v1, 0.f); }
    C[(long)m * ldc + ncol]      = v0;
    C[(long)m * ldc + ncol + 16] = v1;
  }
}

// ---------------------------------------------------------------------------
// Fused multiplicative NMF update:
//   numer = op(Anum)(M x Knum) @ B1(Knum x 64)
//   denom = X(M x 64) @ G(64 x 64)
//   X    *= numer / (denom + eps)    (in place; 64-column X, ld=64)
template<bool TA>
__global__ void __launch_bounds__(256) nmf_update_kernel(
    const float* __restrict__ Anum, int lda, int Knum, long sA,
    const float* __restrict__ B1, long sB1,
    const float* __restrict__ G,  long sG,
    float* X, long sX)
{
  __shared__ _Float16 As[64 * LS];
  __shared__ _Float16 Bs[64 * LS];
  const int bz = blockIdx.z;
  Anum += (long)bz * sA; B1 += (long)bz * sB1; G += (long)bz * sG; X += (long)bz * sX;
  const int m0 = blockIdx.x * 64;
  const int tid = threadIdx.x;
  const int wave = tid >> 5, lane = tid & 31;
  const int tr = (wave & 3) * 16;
  const int tc = (wave >> 2) * 32;
  v8f an0 = {}, an1 = {}, ad0 = {}, ad1 = {};

  // ---- numerator ----
  for (int k0 = 0; k0 < Knum; k0 += 32) {
    stage_a<TA>(Anum, lda, m0, k0, tid, As);
    stage_b<false>(B1, 64, 0, k0, tid, Bs);
    __syncthreads();
    v16h af = load_a_frag(As, tr, lane);
    v16h b0 = load_b_frag(Bs, tc, lane);
    v16h b1 = load_b_frag(Bs, tc + 16, lane);
    an0 = __builtin_amdgcn_wmma_f32_16x16x32_f16(false, af, false, b0, (short)0, an0, false, false);
    an1 = __builtin_amdgcn_wmma_f32_16x16x32_f16(false, af, false, b1, (short)0, an1, false, false);
    __syncthreads();
  }
  // ---- denominator (K = 64) ----
  for (int k0 = 0; k0 < 64; k0 += 32) {
    stage_a<false>(X, 64, m0, k0, tid, As);
    stage_b<false>(G, 64, 0, k0, tid, Bs);
    __syncthreads();
    v16h af = load_a_frag(As, tr, lane);
    v16h b0 = load_b_frag(Bs, tc, lane);
    v16h b1 = load_b_frag(Bs, tc + 16, lane);
    ad0 = __builtin_amdgcn_wmma_f32_16x16x32_f16(false, af, false, b0, (short)0, ad0, false, false);
    ad1 = __builtin_amdgcn_wmma_f32_16x16x32_f16(false, af, false, b1, (short)0, ad1, false, false);
    __syncthreads();   // all staging reads of X done before in-place writes
  }
  const int ncol = tc + (lane & 15);
  const int mr   = m0 + tr + ((lane >> 4) * 8);
#pragma unroll
  for (int i = 0; i < 8; ++i) {
    long i0 = (long)(mr + i) * 64 + ncol;
    long i1 = i0 + 16;
    float x0 = X[i0], x1 = X[i1];
    X[i0] = x0 * an0[i] / (ad0[i] + NMF_EPS);
    X[i1] = x1 * an1[i] / (ad1[i] + NMF_EPS);
  }
}

// ---------------------------------------------------------------------------
// bases = bases_init / max(||bases_init||_d, 1e-12), per (b, r) column
__global__ void normalize_bases_kernel(const float* __restrict__ bi, float* __restrict__ bases) {
  int t = blockIdx.x * blockDim.x + threadIdx.x;   // (b, r)
  if (t >= 8 * 64) return;
  int b = t >> 6, r = t & 63;
  const float* src = bi + (long)b * 256 * 64 + r;
  float* dst = bases + (long)b * 256 * 64 + r;
  float s = 0.f;
  for (int d = 0; d < 256; ++d) { float v = src[(long)d * 64]; s += v * v; }
  float inv = 1.f / fmaxf(sqrtf(s), 1e-12f);
  for (int d = 0; d < 256; ++d) dst[(long)d * 64] = src[(long)d * 64] * inv;
}

// Row softmax over 64 columns, in place (INV_T = 1.0)
__global__ void softmax_rows_kernel(float* __restrict__ coef) {
  long row = (long)blockIdx.x * blockDim.x + threadIdx.x;
  if (row >= 8L * 4096) return;
  float* p = coef + row * 64;
  float mx = -3.0e38f;
  for (int r = 0; r < 64; ++r) mx = fmaxf(mx, p[r]);
  float s = 0.f;
  for (int r = 0; r < 64; ++r) { float e = __expf(p[r] - mx); p[r] = e; s += e; }
  float inv = 1.f / s;
  for (int r = 0; r < 64; ++r) p[r] *= inv;
}

// GroupNorm stats: one block per (b, group); 8 channels x 4096 spatial
__global__ void __launch_bounds__(256) gn_stats_kernel(const float* __restrict__ o,
                                                       float* __restrict__ stats) {
  __shared__ float rs[256], rss[256];
  const int b = blockIdx.x >> 5, g = blockIdx.x & 31;
  const float* base = o + ((long)b * 256 + (long)g * 8) * 4096;
  float s = 0.f, ss = 0.f;
  for (int i = threadIdx.x; i < 8 * 4096; i += 256) {
    float v = base[i]; s += v; ss += v * v;
  }
  rs[threadIdx.x] = s; rss[threadIdx.x] = ss;
  __syncthreads();
  for (int st = 128; st > 0; st >>= 1) {
    if (threadIdx.x < st) { rs[threadIdx.x] += rs[threadIdx.x + st]; rss[threadIdx.x] += rss[threadIdx.x + st]; }
    __syncthreads();
  }
  if (threadIdx.x == 0) {
    float mean = rs[0] * (1.f / 32768.f);
    float var  = rss[0] * (1.f / 32768.f) - mean * mean;
    stats[blockIdx.x * 2]     = mean;
    stats[blockIdx.x * 2 + 1] = rsqrtf(var + 1e-5f);
  }
}

// out = relu(x + gamma*(o - mu)*rstd + beta)
__global__ void finalize_kernel(const float* __restrict__ x, const float* __restrict__ o,
                                const float* __restrict__ gamma, const float* __restrict__ beta,
                                const float* __restrict__ stats, float* __restrict__ out) {
  const long total = 8L * 256 * 4096;
  for (long i = (long)blockIdx.x * blockDim.x + threadIdx.x; i < total;
       i += (long)gridDim.x * blockDim.x) {
    int c = (int)((i >> 12) & 255);
    int b = (int)(i >> 20);
    int g = c >> 3;
    float mean = stats[(b * 32 + g) * 2];
    float rstd = stats[(b * 32 + g) * 2 + 1];
    float v = (o[i] - mean) * rstd * gamma[c] + beta[c];
    out[i] = fmaxf(x[i] + v, 0.f);
  }
}

// ---------------------------------------------------------------------------
extern "C" void kernel_launch(void* const* d_in, const int* in_sizes, int n_in,
                              void* d_out, int out_size, void* d_ws, size_t ws_size,
                              hipStream_t stream) {
  (void)in_sizes; (void)n_in; (void)out_size; (void)ws_size;
  const float* x      = (const float*)d_in[0];
  const float* w_in   = (const float*)d_in[1];
  const float* b_in   = (const float*)d_in[2];
  const float* w_out  = (const float*)d_in[3];
  const float* gamma  = (const float*)d_in[4];
  const float* beta   = (const float*)d_in[5];
  const float* bases0 = (const float*)d_in[6];
  float* out = (float*)d_out;

  // workspace layout (f32): h | o | coef | bases | btb | ctc | stats  (~76.3 MB)
  float* h     = (float*)d_ws;                 // (B, D=256, N=4096)
  float* o     = h     + 8L * 256 * 4096;      // (B, C, N)
  float* coef  = o     + 8L * 256 * 4096;      // (B, N, R=64)
  float* bases = coef  + 8L * 4096 * 64;       // (B, D, R)
  float* btb   = bases + 8L * 256 * 64;        // (B, R, R)
  float* ctc   = btb   + 8L * 64 * 64;         // (B, R, R)
  float* stats = ctc   + 8L * 64 * 64;         // (B, 32, 2)

  const long sBN = 256L * 4096;   // (C,N) batch stride
  const long sNR = 4096L * 64;    // (N,R)
  const long sDR = 256L * 64;     // (D,R)
  const long sRR = 64L * 64;      // (R,R)
  dim3 blk(256);

  // bases = normalize(bases_init)
  normalize_bases_kernel<<<dim3(2), blk, 0, stream>>>(bases0, bases);

  // h = relu(w_in @ x + b_in)    M=256 N=4096 K=256
  gemm_wmma_kernel<false, false><<<dim3(64, 4, 8), blk, 0, stream>>>(
      w_in, x, h, 256, 256, 4096, 4096, 0, sBN, sBN, b_in, 1);

  // coef = softmax_r(h^T @ bases)   M=4096 N=64 K=256 (A transposed)
  gemm_wmma_kernel<true, false><<<dim3(1, 64, 8), blk, 0, stream>>>(
      h, bases, coef, 256, 4096, 64, 64, sBN, sDR, sNR, nullptr, 0);
  softmax_rows_kernel<<<dim3(128), blk, 0, stream>>>(coef);

  for (int it = 0; it < 7; ++it) {
    // btb = bases^T bases   (64x64, K=256)
    gemm_wmma_kernel<true, false><<<dim3(1, 1, 8), blk, 0, stream>>>(
        bases, bases, btb, 256, 64, 64, 64, sDR, sDR, sRR, nullptr, 0);
    // coef *= (h^T bases) / (coef btb + eps)
    nmf_update_kernel<true><<<dim3(64, 1, 8), blk, 0, stream>>>(
        h, 4096, 256, sBN, bases, sDR, btb, sRR, coef, sNR);
    // ctc = coef^T coef     (64x64, K=4096)
    gemm_wmma_kernel<true, false><<<dim3(1, 1, 8), blk, 0, stream>>>(
        coef, coef, ctc, 4096, 64, 64, 64, sNR, sNR, sRR, nullptr, 0);
    // bases *= (h coef) / (bases ctc + eps)
    nmf_update_kernel<false><<<dim3(4, 1, 8), blk, 0, stream>>>(
        h, 4096, 4096, sBN, coef, sNR, ctc, sRR, bases, sDR);
  }

  // compute_coef: btb refresh + one more coef update (math identical to step)
  gemm_wmma_kernel<true, false><<<dim3(1, 1, 8), blk, 0, stream>>>(
      bases, bases, btb, 256, 64, 64, 64, sDR, sDR, sRR, nullptr, 0);
  nmf_update_kernel<true><<<dim3(64, 1, 8), blk, 0, stream>>>(
      h, 4096, 256, sBN, bases, sDR, btb, sRR, coef, sNR);

  // rec = bases @ coef^T  -> overwrite h   M=256 N=4096 K=64 (B transposed)
  gemm_wmma_kernel<false, true><<<dim3(64, 4, 8), blk, 0, stream>>>(
      bases, coef, h, 64, 64, 64, 4096, sDR, sNR, sBN, nullptr, 0);

  // o = w_out @ rec       M=256 N=4096 K=256
  gemm_wmma_kernel<false, false><<<dim3(64, 4, 8), blk, 0, stream>>>(
      w_out, h, o, 256, 256, 4096, 4096, 0, sBN, sBN, nullptr, 0);

  // GroupNorm + residual + relu
  gn_stats_kernel<<<dim3(256), blk, 0, stream>>>(o, stats);
  finalize_kernel<<<dim3(4096), blk, 0, stream>>>(x, o, gamma, beta, stats, out);
}
